// BioConvolution_45561013076214
// MI455X (gfx1250) — matrix operand order repaired and checked by reading, not compiled, not run on hardware
//
#include <hip/hip_runtime.h>
#include <hip/hip_bf16.h>

// Locally-connected conv as 256 independent GEMMs:
//   per patch p: Z_p[64,128] = A_p[64,2048] x B_p[2048,128], bias + relu.
// Roofline: 8.6 GFLOP vs 411 MB streamed once -> AI ~21 FLOP/B, memory floor
// ~17.6 us @ 23.3 TB/s. So: f32->f16 convert in registers + v_wmma_f32_16x16x32_f16
// (f32 accum) keeps the matrix pipe far under the memory floor, and ALL global
// traffic is non-temporal (zero inter-block reuse; don't thrash the 192MB L2).
// K is staged in chunks of 64 (two WMMA K-steps) through double-buffered LDS.

typedef __attribute__((ext_vector_type(16))) _Float16 v16h;
typedef __attribute__((ext_vector_type(8)))  _Float16 v8h;
typedef __attribute__((ext_vector_type(4)))  _Float16 v4h;
typedef __attribute__((ext_vector_type(8)))  float    v8f;
typedef __attribute__((ext_vector_type(4)))  float    v4f;

#define N_IMG  64
#define IMG_H  128
#define IMG_W  128
#define FIN    32
#define FH     8
#define FW     8
#define FOUT   128
#define KTOT   2048              // FH*FW*FIN
#define KB     64                // K per staged chunk == two WMMA K-steps
#define NCHUNK (KTOT / KB)       // 32
#define BROW   72                // padded K-stride of transposed B tile (halfs); 144B rows, 16B-aligned

__global__ __launch_bounds__(256)
void BioConvolution_lc_wmma_kernel(const float* __restrict__ X,
                                   const float* __restrict__ Fw,
                                   const float* __restrict__ bias,
                                   float* __restrict__ Out)
{
    // Double-buffered LDS: A (64x64 f16) row-major, B (64x128 f16) transposed.
    __shared__ _Float16 Ash[2][N_IMG][KB];     // 16 KB
    __shared__ _Float16 BshT[2][FOUT][BROW];   // 36 KB

    const int p    = blockIdx.x;              // patch 0..255
    const int pr   = p >> 4;                  // patch row (16x16 patch grid)
    const int pc   = p & 15;
    const int t    = threadIdx.x;
    const int lane = t & 31;
    const int wave = t >> 5;                  // 8 waves

    const float* Fp = Fw + (size_t)p * KTOT * FOUT;

    // ---- staging thread mapping ----
    // A: K chunk of 64 = two adjacent filter-x pixels => 64 CONTIGUOUS floats
    //    per image row ([W][FIN] layout). thread -> (image, 16-float slice).
    const int an  = t >> 2;                   // image 0..63
    const int ac  = (t & 3) * 16;             // slice within the 64 K values
    // B: thread -> (4 k-rows, 8 fout cols); stores transposed as half4 {k..k+3}
    const int bk  = (t >> 4) * 4;             // k base 0,4,..,60
    const int bn0 = (t & 15) * 8;             // fout base

    // ---- compute mapping: wave -> 16x64 strip of the 64x128 output ----
    const int mt    = wave & 3;               // M tile (images 16*mt..)
    const int nh    = wave >> 2;              // N half (fout 64*nh..)
    const int mrow  = (mt << 4) + (lane & 15);
    const int kbA   = (lane >> 4) * 8;        // f16 A-frag K base (ISA layout)
    const int kbB   = (lane >> 4) * 8;        // f16 B-frag K base (ISA layout)
    const int ncol0 = (nh << 6) + (lane & 15);

    v8f acc[4] = {v8f{}, v8f{}, v8f{}, v8f{}};

    auto stage = [&](int kc, int buf) {
        const int i  = kc >> 2;               // filter row
        const int j2 = (kc & 3) * 2;          // first of two filter cols
        // ---- A: 64 contiguous floats (2 pixels x 32 ch) per image ----
        const float* ap = X + (((size_t)an * IMG_H + (pr * FH + i)) * IMG_W
                               + (pc * FW + j2)) * FIN + ac;
        v4f a0 = __builtin_nontemporal_load((const v4f*)(ap));
        v4f a1 = __builtin_nontemporal_load((const v4f*)(ap + 4));
        v4f a2 = __builtin_nontemporal_load((const v4f*)(ap + 8));
        v4f a3 = __builtin_nontemporal_load((const v4f*)(ap + 12));
        v8h ah0, ah1;
#pragma unroll
        for (int u = 0; u < 4; ++u) {
            ah0[u]     = (_Float16)a0[u];
            ah0[4 + u] = (_Float16)a1[u];
            ah1[u]     = (_Float16)a2[u];
            ah1[4 + u] = (_Float16)a3[u];
        }
        *(v8h*)&Ash[buf][an][ac]     = ah0;   // 16B ds_store
        *(v8h*)&Ash[buf][an][ac + 8] = ah1;

        // ---- B: rows kc*64+bk .. +3, cols bn0..bn0+7; store transposed ----
        const float* bp = Fp + (size_t)(kc * KB + bk) * FOUT + bn0;
        v4f brow[4][2];
#pragma unroll
        for (int r = 0; r < 4; ++r) {
            brow[r][0] = __builtin_nontemporal_load((const v4f*)(bp + r * FOUT));
            brow[r][1] = __builtin_nontemporal_load((const v4f*)(bp + r * FOUT + 4));
        }
#pragma unroll
        for (int u = 0; u < 8; ++u) {
            v4h h4;
#pragma unroll
            for (int r = 0; r < 4; ++r)
                h4[r] = (_Float16)brow[r][u >> 2][u & 3];
            *(v4h*)&BshT[buf][bn0 + u][bk] = h4;  // 8B ds_store
        }
    };

    auto compute = [&](int buf) {
#pragma unroll
        for (int kt = 0; kt < 2; ++kt) {      // two WMMA K-steps per chunk
            const int ka = kt * 32;
            // A fragment: two contiguous 16B runs per lane (ISA 16-bit A layout).
            v8h alo = *(const v8h*)&Ash[buf][mrow][ka + kbA];
            v8h ahi = *(const v8h*)&Ash[buf][mrow][ka + 16 + kbA];
            v16h a  = __builtin_shufflevector(alo, ahi,
                        0,1,2,3,4,5,6,7,8,9,10,11,12,13,14,15);
#pragma unroll
            for (int q = 0; q < 4; ++q) {
                const int n = ncol0 + (q << 4);
                v8h blo = *(const v8h*)&BshT[buf][n][ka + kbB];
                v8h bhi = *(const v8h*)&BshT[buf][n][ka + kbB + 16];
                v16h b  = __builtin_shufflevector(blo, bhi,
                            0,1,2,3,4,5,6,7,8,9,10,11,12,13,14,15);
                acc[q] = __builtin_amdgcn_wmma_f32_16x16x32_f16(
                             /*neg_a=*/false, a, /*neg_b=*/false, b,
                             /*c_mod=*/(short)0, acc[q],
                             /*reuse_a=*/false, /*reuse_b=*/false);
            }
        }
    };

    // ---- software-pipelined K loop (double-buffered, 1 barrier/chunk) ----
    stage(0, 0);
    __syncthreads();
    for (int kc = 0; kc < NCHUNK; ++kc) {
        const int buf = kc & 1;
        if (kc + 1 < NCHUNK) stage(kc + 1, buf ^ 1);
        compute(buf);
        __syncthreads();
    }

    // ---- epilogue: bias + relu, per documented C/D layout ----
    // VGPR r, lanes 0-15: M=r ; lanes 16-31: M=8+r ; N = lane&15 (per tile)
#pragma unroll
    for (int q = 0; q < 4; ++q) {
        const int ncol = ncol0 + (q << 4);
        const float bv = bias[ncol];
#pragma unroll
        for (int r = 0; r < 8; ++r) {
            const int m = (mt << 4) + ((lane >> 4) << 3) + r;   // image index
            float v = acc[q][r] + bv;
            v = v > 0.0f ? v : 0.0f;
            __builtin_nontemporal_store(v, &Out[((size_t)m * 256 + p) * FOUT + ncol]);
        }
    }
}

extern "C" void kernel_launch(void* const* d_in, const int* in_sizes, int n_in,
                              void* d_out, int out_size, void* d_ws, size_t ws_size,
                              hipStream_t stream) {
    (void)in_sizes; (void)n_in; (void)out_size; (void)d_ws; (void)ws_size;
    const float* X    = (const float*)d_in[0];   // [64,128,128,32] f32
    const float* Fw   = (const float*)d_in[1];   // [256,8,8,32,128] f32
    const float* bias = (const float*)d_in[2];   // [128] f32
    float*       Out  = (float*)d_out;           // [64,16,16,128] f32

    dim3 grid(256);   // one workgroup per patch
    dim3 block(256);  // 8 wave32
    BioConvolution_lc_wmma_kernel<<<grid, block, 0, stream>>>(X, Fw, bias, Out);
}
